// Reinforced_GATLayer_4621384810787
// MI455X (gfx1250) — matrix operand order repaired and checked by reading, not compiled, not run on hardware
//
#include <hip/hip_runtime.h>
#include <hip/hip_bf16.h>
#include <math.h>

#define GN    8192
#define DIN   256
#define DOUT  128
#define LRELU 0.2f
#define NEG_BIG (-9.0e15f)

typedef __attribute__((ext_vector_type(16))) _Float16 v16h;
typedef __attribute__((ext_vector_type(8)))  float    v8f;

// ---------------------------------------------------------------------------
// Kernel 1: H = X @ W  via v_wmma_f32_16x16x32_f16 (f16 in, f32 accumulate).
// Grid 64 x 256 threads (8 waves). Wave w computes rows [blk*128+w*16, +16)
// across all 128 output columns (8 tiles of 16x16, K-loop of 8 x 32).
//
// W is staged in LDS *pre-swizzled into WMMA B-fragment order*:
//   wfrag[kstep][ct][lane] = 16 contiguous halves (32B) = that lane's full
//   B fragment for (kstep, col-tile ct). Compute loop then does one aligned
//   v16h LDS read per fragment (ds_load_b128 x2) with no half-repacking.
// Fragment element e (0..15) of lane L holds:
//   K = kstep*32 + (L>=16 ? 16 : 0) + e,  N = ct*16 + (L&15)
// ---------------------------------------------------------------------------
__global__ __launch_bounds__(256) void gemm_h(const float* __restrict__ X,
                                              const float* __restrict__ W,
                                              float* __restrict__ H) {
  __shared__ v16h wfrag[8 * 8 * 32];          // 64 KB, fragment-major
  _Float16* wh = (_Float16*)wfrag;

  const int tid = threadIdx.x;
  // Bijective staging permutation: idx -> (kstep, ct, lane, elem)
  for (int idx = tid; idx < DIN * DOUT; idx += 256) {
    const int elem  = idx & 15;
    const int lane  = (idx >> 4) & 31;
    const int ct    = (idx >> 9) & 7;
    const int kstep = idx >> 12;
    const int K = kstep * 32 + ((lane >> 4) << 4) + elem;
    const int Ncol = ct * 16 + (lane & 15);
    wh[idx] = (_Float16)W[K * DOUT + Ncol];
  }
  __syncthreads();

  const int wid    = tid >> 5;
  const int lane   = tid & 31;
  const int laneHi = (lane >> 4) & 1;   // 0: lanes 0-15, 1: lanes 16-31
  const int m      = lane & 15;         // A: row within tile / B,C,D: col
  const int rowBase = blockIdx.x * 128 + wid * 16;

  v8f acc[8];
#pragma unroll
  for (int ct = 0; ct < 8; ++ct) acc[ct] = (v8f){0.f,0.f,0.f,0.f,0.f,0.f,0.f,0.f};

#pragma unroll
  for (int ks8 = 0; ks8 < 8; ++ks8) {
    // A fragment (16x32 f16): lane holds row m; VGPR r halves h:
    //   K = laneHi*16 + 2*r + h   (contiguous pairs -> float2)
    v16h a;
    const float* xrow = X + (size_t)(rowBase + m) * DIN + ks8 * 32;
#pragma unroll
    for (int r = 0; r < 8; ++r) {
      const int k0 = laneHi * 16 + 2 * r;
      const float2 v = *(const float2*)(xrow + k0);
      a[2 * r]     = (_Float16)v.x;
      a[2 * r + 1] = (_Float16)v.y;
    }
#pragma unroll
    for (int ct = 0; ct < 8; ++ct) {
      const v16h b = wfrag[(ks8 * 8 + ct) * 32 + lane];   // one 32B LDS read
      acc[ct] = __builtin_amdgcn_wmma_f32_16x16x32_f16(
          false, a, false, b, (short)0, acc[ct], false, false);
    }
  }
  // C/D layout: VGPR v -> row M = v + laneHi*8, col = ct*16 + m
#pragma unroll
  for (int ct = 0; ct < 8; ++ct) {
#pragma unroll
    for (int v = 0; v < 8; ++v) {
      const int M = v + laneHi * 8;
      H[(size_t)(rowBase + M) * DOUT + ct * 16 + m] = acc[ct][v];
    }
  }
}

// ---------------------------------------------------------------------------
// Kernel 2: s[i] = h[i]·a_self, t[i] = h[i]·a_neighs  (one thread per row)
// ---------------------------------------------------------------------------
__global__ void attn_dots(const float* __restrict__ H,
                          const float* __restrict__ a_self,
                          const float* __restrict__ a_neighs,
                          float* __restrict__ s, float* __restrict__ t) {
  const int row = blockIdx.x * blockDim.x + threadIdx.x;
  if (row >= GN) return;
  const float* hr = H + (size_t)row * DOUT;
  float ss = 0.f, tt = 0.f;
#pragma unroll 4
  for (int c = 0; c < DOUT; ++c) {
    const float hv = hr[c];
    ss += hv * a_self[c];
    tt += hv * a_neighs[c];
  }
  s[row] = ss;
  t[row] = tt;
}

// ---------------------------------------------------------------------------
// Kernel 3: per-row masked softmax + top-(action+1) threshold + sparse
// gather-accumulate of h, then ELU. One 256-thread block per row.
// The adj row (32KB) is the dominant HBM traffic; stream it once with an
// L2 prefetch ahead of the consume point (global_prefetch_b8).
// ---------------------------------------------------------------------------
__global__ __launch_bounds__(256) void gat_row(const float* __restrict__ H,
                                               const int*   __restrict__ adj,
                                               const int*   __restrict__ action,
                                               const float* __restrict__ s,
                                               const float* __restrict__ t,
                                               const int*   __restrict__ breadth,
                                               float* __restrict__ out) {
  __shared__ float e_row[GN];     // 32 KB masked scores
  __shared__ float red[256];
  __shared__ float accs[8 * DOUT];

  const int i   = blockIdx.x;
  const int tid = threadIdx.x;
  const float si = s[i];
  const int* arow = adj + (size_t)i * GN;

  // masked e + row max
  float lmax = -INFINITY;
  for (int j = tid; j < GN; j += 256) {
    if (j + 2048 < GN)
      __builtin_prefetch(arow + j + 2048, 0, 1);   // global_prefetch_b8
    float ev;
    if (arow[j] > 0) {
      const float x = si + t[j];
      ev = (x >= 0.f) ? x : LRELU * x;
    } else {
      ev = NEG_BIG;
    }
    e_row[j] = ev;
    lmax = fmaxf(lmax, ev);
  }
  red[tid] = lmax;
  __syncthreads();
  for (int off = 128; off > 0; off >>= 1) {
    if (tid < off) red[tid] = fmaxf(red[tid], red[tid + off]);
    __syncthreads();
  }
  const float rowmax = red[0];
  __syncthreads();

  // softmax denominator (masked entries underflow to exactly 0)
  float lsum = 0.f;
  for (int j = tid; j < GN; j += 256) lsum += expf(e_row[j] - rowmax);
  red[tid] = lsum;
  __syncthreads();
  for (int off = 128; off > 0; off >>= 1) {
    if (tid < off) red[tid] += red[tid + off];
    __syncthreads();
  }
  const float denom = red[0];
  __syncthreads();

  // threshold = (action[i]+1)-th largest e (softmax is monotone in e):
  // action[i]+1 strict-max extraction passes over the LDS-resident row.
  float thr = -INFINITY;
  if (breadth[0] != 0) {
    thr = INFINITY;
    const int k = action[i];
    for (int itr = 0; itr <= k; ++itr) {
      float lm = -INFINITY;
      for (int j = tid; j < GN; j += 256) {
        const float ev = e_row[j];
        if (ev < thr) lm = fmaxf(lm, ev);
      }
      red[tid] = lm;
      __syncthreads();
      for (int off = 128; off > 0; off >>= 1) {
        if (tid < off) red[tid] = fmaxf(red[tid], red[tid + off]);
        __syncthreads();
      }
      thr = red[0];
      __syncthreads();
    }
  }

  // sparse accumulate: ~action[i]+1 surviving columns per row.
  const int wid = tid >> 5, lane = tid & 31;
  float a0 = 0.f, a1 = 0.f, a2 = 0.f, a3 = 0.f;
  const int jBeg = wid * (GN / 8), jEnd = jBeg + (GN / 8);
  for (int jb = jBeg; jb < jEnd; jb += 32) {
    const float ev = e_row[jb + lane];
    const float w  = (ev >= thr) ? expf(ev - rowmax) : 0.f;
    unsigned mask = (unsigned)__ballot(w > 0.f);
    while (mask) {
      const int b = __ffs(mask) - 1;
      mask &= mask - 1;
      const float wj = __shfl(w, b, 32);
      const float* hj = H + (size_t)(jb + b) * DOUT;
      a0 += wj * hj[lane];
      a1 += wj * hj[lane + 32];
      a2 += wj * hj[lane + 64];
      a3 += wj * hj[lane + 96];
    }
  }
  accs[wid * DOUT + lane]      = a0;
  accs[wid * DOUT + lane + 32] = a1;
  accs[wid * DOUT + lane + 64] = a2;
  accs[wid * DOUT + lane + 96] = a3;
  __syncthreads();

  if (tid < DOUT) {
    float v = 0.f;
#pragma unroll
    for (int w8 = 0; w8 < 8; ++w8) v += accs[w8 * DOUT + tid];
    v /= denom;
    out[(size_t)i * DOUT + tid] = (v > 0.f) ? v : expm1f(v);  // ELU
  }
}

// ---------------------------------------------------------------------------
extern "C" void kernel_launch(void* const* d_in, const int* in_sizes, int n_in,
                              void* d_out, int out_size, void* d_ws, size_t ws_size,
                              hipStream_t stream) {
  const float* X        = (const float*)d_in[0];
  const int*   adj      = (const int*)  d_in[1];
  const int*   action   = (const int*)  d_in[2];
  const float* W        = (const float*)d_in[3];
  const float* a_self   = (const float*)d_in[4];
  const float* a_neighs = (const float*)d_in[5];
  const int*   breadth  = (const int*)  d_in[6];
  float*       out      = (float*)d_out;

  float* H = (float*)d_ws;                 // [GN, DOUT]  4 MB
  float* s = H + (size_t)GN * DOUT;        // [GN]
  float* t = s + GN;                       // [GN]

  gemm_h   <<<dim3(GN / 128), dim3(256), 0, stream>>>(X, W, H);
  attn_dots<<<dim3(GN / 256), dim3(256), 0, stream>>>(H, a_self, a_neighs, s, t);
  gat_row  <<<dim3(GN),       dim3(256), 0, stream>>>(H, adj, action, s, t, breadth, out);
}
